// AdvancedCardiacGNN_28475633172515
// MI455X (gfx1250) — matrix operand order repaired and checked by reading, not compiled, not run on hardware
//
#include <hip/hip_runtime.h>
#include <hip/hip_bf16.h>
#include <math.h>

// ---------------------------------------------------------------------------
// CDNA5 (gfx1250) implementation of AdvancedCardiacGNN forward pass.
// Dense GEMMs: V_WMMA_F32_16X16X32_BF16 (wave32). Tiles are staged into LDS
// in *fragment order* so each lane reads its whole 16-element bf16 fragment
// with two ds_load_b128s. Global loads are coalesced float4 / b32 with a
// fast path (no guards) for interior tiles.
// ---------------------------------------------------------------------------

typedef __attribute__((ext_vector_type(16))) __bf16 v16bf;
typedef __attribute__((ext_vector_type(4)))  __bf16 v4bf;
typedef __attribute__((ext_vector_type(8)))  float  v8f;

#define N_NODES 10000
#define N_EDGES 320000
#define N_GRAPHS 8
#define HEADS 4

// ---------------------------------------------------------------------------
// GEMM: Y[M,N] = act(A[M,K] @ W[K,N] + bias[N]); row-major fp32 in, fp32 out.
// Block: 128 threads = 4 waves. Block tile: 32(M) x 64(N) x 32(K).
// Each wave owns one 16-wide N column group and computes two 16x16 C tiles
// (two M subtiles) per K step -> 2 static v_wmma per loop.
//
// LDS layout (fragment order, ISA 7.12.2 16-bit layouts, wave32):
//   A: lane = r + 16*((k>>3)&1), elem e = (k&7) + ((k>>4)<<3)
//   B: lane = (n&15) + 16*((k>>3)&1), elem e = (k&7) + ((k>>4)<<3),  per 16-col group w = n>>4
// act: 0 = identity, 1 = relu. Requires N % 16 == 0.
// ---------------------------------------------------------------------------
__global__ __launch_bounds__(128)
void gemm_bias_act_wmma(const float* __restrict__ A, const float* __restrict__ W,
                        const float* __restrict__ bias, float* __restrict__ Y,
                        int M, int K, int N, int act)
{
    __shared__ __bf16 Asf[2][32][16];   // 2 KB: two 16-row M subtiles
    __shared__ __bf16 Bsf[4][32][16];   // 4 KB: four 16-col N groups

    const int tid    = threadIdx.x;
    const int lane   = tid & 31;
    const int wave   = tid >> 5;              // 0..3 -> N group
    const int mBase  = blockIdx.x * 32;
    const int nBlock = blockIdx.y * 64;

    v8f acc0 = {};
    v8f acc1 = {};

    for (int k0 = 0; k0 < K; k0 += 32) {
        const bool fast = (k0 + 32 <= K) && (mBase + 32 <= M) && (nBlock + 64 <= N);

        // ---- A tile (32 rows x 32 k): 256 float4 / 128 threads = 2 iters ----
        #pragma unroll
        for (int it = 0; it < 2; ++it) {
            int i    = tid + it * 128;        // 0..255
            int sub  = i >> 7;                // M subtile
            int r    = (i >> 3) & 15;         // row in subtile
            int c4   = (i & 7) * 4;           // k offset (multiple of 4)
            int gr   = mBase + sub * 16 + r;
            int gk   = k0 + c4;
            float4 a;
            if (fast) {
                a = *(const float4*)(A + (size_t)gr * K + gk);
            } else {
                a.x = (gr < M && gk + 0 < K) ? A[(size_t)gr * K + gk + 0] : 0.0f;
                a.y = (gr < M && gk + 1 < K) ? A[(size_t)gr * K + gk + 1] : 0.0f;
                a.z = (gr < M && gk + 2 < K) ? A[(size_t)gr * K + gk + 2] : 0.0f;
                a.w = (gr < M && gk + 3 < K) ? A[(size_t)gr * K + gk + 3] : 0.0f;
            }
            int dlane = r + 16 * ((c4 >> 3) & 1);
            int e0    = (c4 & 7) + ((c4 >> 4) << 3);   // multiple of 4 -> 8B aligned
            v4bf p = { (__bf16)a.x, (__bf16)a.y, (__bf16)a.z, (__bf16)a.w };
            *(v4bf*)&Asf[sub][dlane][e0] = p;
        }

        // ---- B tile (32 k x 64 n): lane-coalesced b32 loads, 4 iters ----
        #pragma unroll
        for (int it = 0; it < 4; ++it) {
            int col = tid & 63;                       // 0..63
            int kq  = (tid >> 6) + it * 2;            // 0..7 -> k quad
            int gn  = nBlock + col;
            float4 b;
            if (fast) {
                const float* wp = W + (size_t)(k0 + kq * 4) * N + gn;
                b.x = wp[0 * (size_t)N];
                b.y = wp[1 * (size_t)N];
                b.z = wp[2 * (size_t)N];
                b.w = wp[3 * (size_t)N];
            } else {
                int gk = k0 + kq * 4;
                b.x = (gk + 0 < K && gn < N) ? W[(size_t)(gk + 0) * N + gn] : 0.0f;
                b.y = (gk + 1 < K && gn < N) ? W[(size_t)(gk + 1) * N + gn] : 0.0f;
                b.z = (gk + 2 < K && gn < N) ? W[(size_t)(gk + 2) * N + gn] : 0.0f;
                b.w = (gk + 3 < K && gn < N) ? W[(size_t)(gk + 3) * N + gn] : 0.0f;
            }
            int w     = col >> 4;
            int dlane = (col & 15) + 16 * (((kq * 4) >> 3) & 1);
            int e0    = ((kq * 4) & 7) + (((kq * 4) >> 4) << 3);   // multiple of 4 -> 8B aligned
            v4bf p = { (__bf16)b.x, (__bf16)b.y, (__bf16)b.z, (__bf16)b.w };
            *(v4bf*)&Bsf[w][dlane][e0] = p;
        }
        __syncthreads();

        // ---- fragments: contiguous 32B per lane -> 2x ds_load_b128 each ----
        v16bf a0 = *(const v16bf*)&Asf[0][lane][0];
        v16bf a1 = *(const v16bf*)&Asf[1][lane][0];
        v16bf bf = *(const v16bf*)&Bsf[wave][lane][0];

        acc0 = __builtin_amdgcn_wmma_f32_16x16x32_bf16(false, a0, false, bf,
                                                       (short)0, acc0, false, false);
        acc1 = __builtin_amdgcn_wmma_f32_16x16x32_bf16(false, a1, false, bf,
                                                       (short)0, acc1, false, false);
        __syncthreads();
    }

    // ---- store: C/D layout: VGPR r -> M = r + 8*(lane/16), N = lane%16 ----
    const int col = nBlock + wave * 16 + (lane & 15);
    if (col < N) {
        const float bv = bias[col];
        const int rowOff = (lane >> 4) * 8;
        #pragma unroll
        for (int r = 0; r < 8; ++r) {
            int row0 = mBase + rowOff + r;
            if (row0 < M) {
                float y = acc0[r] + bv;
                if (act == 1) y = fmaxf(y, 0.0f);
                Y[(size_t)row0 * N + col] = y;
            }
            int row1 = mBase + 16 + rowOff + r;
            if (row1 < M) {
                float y = acc1[r] + bv;
                if (act == 1) y = fmaxf(y, 0.0f);
                Y[(size_t)row1 * N + col] = y;
            }
        }
    }
}

// ---------------------------------------------------------------------------
// Utility / edge kernels
// ---------------------------------------------------------------------------
__global__ void fill_f32(float* __restrict__ p, float v, long long n)
{
    long long stride = (long long)gridDim.x * blockDim.x;
    for (long long i = (long long)blockIdx.x * blockDim.x + threadIdx.x; i < n; i += stride)
        p[i] = v;
}

__device__ inline void atomicMaxF(float* addr, float val)
{
    unsigned* ua  = (unsigned*)addr;
    unsigned  old = __hip_atomic_load(ua, __ATOMIC_RELAXED, __HIP_MEMORY_SCOPE_AGENT);
    while (__uint_as_float(old) < val) {
        unsigned assumed = old;
        old = atomicCAS(ua, assumed, __float_as_uint(val));
        if (old == assumed) break;
    }
}

// alpha[e,h] = (q[dst] . k[src]) * scale ; amax[dst,h] = segment max
__global__ void edge_logits(const float* __restrict__ q, const float* __restrict__ k,
                            const int* __restrict__ src, const int* __restrict__ dst,
                            float* __restrict__ alpha, float* __restrict__ amax,
                            int E, int C, float scale)
{
    long long total  = (long long)E * HEADS;
    long long stride = (long long)gridDim.x * blockDim.x;
    for (long long i = (long long)blockIdx.x * blockDim.x + threadIdx.x; i < total; i += stride) {
        int e = (int)(i / HEADS);
        int h = (int)(i % HEADS);
        int s = src[e], d = dst[e];
        const float* qp = q + ((size_t)d * HEADS + h) * C;
        const float* kp = k + ((size_t)s * HEADS + h) * C;
        float acc = 0.0f;
        for (int c = 0; c < C; c += 4) {
            float4 qa = *(const float4*)(qp + c);
            float4 ka = *(const float4*)(kp + c);
            acc += qa.x * ka.x + qa.y * ka.y + qa.z * ka.z + qa.w * ka.w;
        }
        acc *= scale;
        alpha[i] = acc;
        atomicMaxF(&amax[(size_t)d * HEADS + h], acc);
    }
}

// alpha[e,h] = exp(alpha - amax[dst,h]) ; denom[dst,h] += alpha
__global__ void edge_expsum(float* __restrict__ alpha, const float* __restrict__ amax,
                            float* __restrict__ denom, const int* __restrict__ dst, int E)
{
    long long total  = (long long)E * HEADS;
    long long stride = (long long)gridDim.x * blockDim.x;
    for (long long i = (long long)blockIdx.x * blockDim.x + threadIdx.x; i < total; i += stride) {
        int e = (int)(i / HEADS);
        int h = (int)(i % HEADS);
        int d = dst[e];
        float ea = __expf(alpha[i] - amax[(size_t)d * HEADS + h]);
        alpha[i] = ea;
        atomicAdd(&denom[(size_t)d * HEADS + h], ea);
    }
}

// alpha[e,h] /= (denom[dst,h] + 1e-16)
__global__ void edge_normalize(float* __restrict__ alpha, const float* __restrict__ denom,
                               const int* __restrict__ dst, int E)
{
    long long total  = (long long)E * HEADS;
    long long stride = (long long)gridDim.x * blockDim.x;
    for (long long i = (long long)blockIdx.x * blockDim.x + threadIdx.x; i < total; i += stride) {
        int e = (int)(i / HEADS);
        int h = (int)(i % HEADS);
        alpha[i] = alpha[i] / (denom[(size_t)dst[e] * HEADS + h] + 1e-16f);
    }
}

// agg[dst,h,c] += v[src,h,c] * alpha[e,h]
__global__ void edge_scatter(const float* __restrict__ v, const float* __restrict__ alpha,
                             const int* __restrict__ src, const int* __restrict__ dst,
                             float* __restrict__ agg, int E, int C)
{
    long long total  = (long long)E * HEADS * C;
    long long stride = (long long)gridDim.x * blockDim.x;
    for (long long i = (long long)blockIdx.x * blockDim.x + threadIdx.x; i < total; i += stride) {
        int c = (int)(i % C);
        long long eh = i / C;
        int h = (int)(eh % HEADS);
        int e = (int)(eh / HEADS);
        atomicAdd(&agg[((size_t)dst[e] * HEADS + h) * C + c],
                  v[((size_t)src[e] * HEADS + h) * C + c] * alpha[eh]);
    }
}

// out[n,c] = relu(LN(mean_h agg[n,h,c] + skip[n,c])) + resid[n,c]
// one block per node, blockDim.x == C (32/64/128, power of two)
__global__ void heads_mean_ln(const float* __restrict__ agg, const float* __restrict__ skip,
                              const float* __restrict__ resid, const float* __restrict__ ln_g,
                              const float* __restrict__ ln_b, float* __restrict__ out, int C)
{
    __shared__ float red[128];
    const int n = blockIdx.x;
    const int c = threadIdx.x;

    float m = 0.0f;
    #pragma unroll
    for (int h = 0; h < HEADS; ++h) m += agg[((size_t)n * HEADS + h) * C + c];
    m = m * (1.0f / HEADS) + skip[(size_t)n * C + c];

    red[c] = m; __syncthreads();
    for (int s = C >> 1; s > 0; s >>= 1) { if (c < s) red[c] += red[c + s]; __syncthreads(); }
    float mu = red[0] * (1.0f / C); __syncthreads();

    float d = m - mu;
    red[c] = d * d; __syncthreads();
    for (int s = C >> 1; s > 0; s >>= 1) { if (c < s) red[c] += red[c + s]; __syncthreads(); }
    float var = red[0] * (1.0f / C); __syncthreads();

    float y = d * rsqrtf(var + 1e-5f) * ln_g[c] + ln_b[c];
    y = fmaxf(y, 0.0f);
    out[(size_t)n * C + c] = y + resid[(size_t)n * C + c];
}

// a[n] = sigmoid(s1[n,:] @ att_W2 + b2) ; g[batch[n], c] += h[n,c] * a[n]
__global__ void pool_attention(const float* __restrict__ s1, const float* __restrict__ W2,
                               const float* __restrict__ b2, const float* __restrict__ h,
                               const int* __restrict__ batch, float* __restrict__ g, int C)
{
    int stride = gridDim.x * blockDim.x;
    for (int n = blockIdx.x * blockDim.x + threadIdx.x; n < N_NODES; n += stride) {
        float s = b2[0];
        #pragma unroll
        for (int j = 0; j < 16; ++j) s += s1[(size_t)n * 16 + j] * W2[j];
        float a = 1.0f / (1.0f + __expf(-s));
        int b = batch[n];
        for (int c = 0; c < C; ++c)
            atomicAdd(&g[(size_t)b * C + c], h[(size_t)n * C + c] * a);
    }
}

// classifier MLP: g[8,32] -> relu 16 -> relu 8 -> 5. Single block.
__global__ void classifier(const float* __restrict__ g,
                           const float* __restrict__ W1, const float* __restrict__ b1,
                           const float* __restrict__ W2, const float* __restrict__ b2,
                           const float* __restrict__ W3, const float* __restrict__ b3,
                           float* __restrict__ out)
{
    __shared__ float z1[8 * 16];
    __shared__ float z2[8 * 8];
    const int t = threadIdx.x;
    if (t < 128) {
        int i = t / 16, j = t % 16;
        float s = b1[j];
        for (int c = 0; c < 32; ++c) s += g[i * 32 + c] * W1[c * 16 + j];
        z1[t] = fmaxf(s, 0.0f);
    }
    __syncthreads();
    if (t < 64) {
        int i = t / 8, j = t % 8;
        float s = b2[j];
        for (int c = 0; c < 16; ++c) s += z1[i * 16 + c] * W2[c * 8 + j];
        z2[t] = fmaxf(s, 0.0f);
    }
    __syncthreads();
    if (t < 40) {
        int i = t / 5, j = t % 5;
        float s = b3[j];
        for (int c = 0; c < 8; ++c) s += z2[i * 8 + c] * W3[c * 5 + j];
        out[t] = s;
    }
}

// ---------------------------------------------------------------------------
// Host-side orchestration
// ---------------------------------------------------------------------------
static inline void launch_gemm(const float* A, const float* W, const float* b, float* Y,
                               int M, int K, int N, int act, hipStream_t stream)
{
    dim3 grid((M + 31) / 32, (N + 63) / 64);
    gemm_bias_act_wmma<<<grid, 128, 0, stream>>>(A, W, b, Y, M, K, N, act);
}

extern "C" void kernel_launch(void* const* d_in, const int* in_sizes, int n_in,
                              void* d_out, int out_size, void* d_ws, size_t ws_size,
                              hipStream_t stream)
{
    (void)in_sizes; (void)n_in; (void)out_size; (void)ws_size;

    // ---- inputs (dict order: x, edge_index, batch, params-as-sorted-pytree) ----
    const float* x     = (const float*)d_in[0];    // [10000, 2000]
    const int*   ei    = (const int*)  d_in[1];    // [2, 320000]
    const int*   batch = (const int*)  d_in[2];    // [10000]
    const int* src = ei;            // edge_index[0]
    const int* dst = ei + N_EDGES;  // edge_index[1]

    // params, flattened with dict keys ASCII-sorted at each level
    const float* att_W1 = (const float*)d_in[3];   // [32,16]
    const float* att_W2 = (const float*)d_in[4];   // [16,1]
    const float* att_b1 = (const float*)d_in[5];   // [16]
    const float* att_b2 = (const float*)d_in[6];   // [1]
    const float* cls_W1 = (const float*)d_in[7];   // [32,16]
    const float* cls_W2 = (const float*)d_in[8];   // [16,8]
    const float* cls_W3 = (const float*)d_in[9];   // [8,5]
    const float* cls_b1 = (const float*)d_in[10];  // [16]
    const float* cls_b2 = (const float*)d_in[11];  // [8]
    const float* cls_b3 = (const float*)d_in[12];  // [5]
    const float* in_W   = (const float*)d_in[13];  // [2000,128]
    const float* in_b   = (const float*)d_in[14];  // [128]

    struct Layer {
        const float *Wq, *bq, *Wk, *bk, *Wv, *bv, *Ws, *bs, *lng, *lnb, *rW, *rb;
        int prev, C;
    } L[3];
    // layer dict key order: Wk, Wq, Wskip, Wv, bk, bq, bskip, bv, ln_b, ln_g, [res_W, res_b]
    {   // layer 0: 128 -> 128, no residual projection
        int b = 15;
        L[0] = { (const float*)d_in[b+1], (const float*)d_in[b+5],   // Wq, bq
                 (const float*)d_in[b+0], (const float*)d_in[b+4],   // Wk, bk
                 (const float*)d_in[b+3], (const float*)d_in[b+7],   // Wv, bv
                 (const float*)d_in[b+2], (const float*)d_in[b+6],   // Wskip, bskip
                 (const float*)d_in[b+9], (const float*)d_in[b+8],   // ln_g, ln_b
                 nullptr, nullptr, 128, 128 };
    }
    {   // layer 1: 128 -> 64, residual projection
        int b = 25;
        L[1] = { (const float*)d_in[b+1], (const float*)d_in[b+5],
                 (const float*)d_in[b+0], (const float*)d_in[b+4],
                 (const float*)d_in[b+3], (const float*)d_in[b+7],
                 (const float*)d_in[b+2], (const float*)d_in[b+6],
                 (const float*)d_in[b+9], (const float*)d_in[b+8],
                 (const float*)d_in[b+10], (const float*)d_in[b+11], 128, 64 };
    }
    {   // layer 2: 64 -> 32, residual projection
        int b = 37;
        L[2] = { (const float*)d_in[b+1], (const float*)d_in[b+5],
                 (const float*)d_in[b+0], (const float*)d_in[b+4],
                 (const float*)d_in[b+3], (const float*)d_in[b+7],
                 (const float*)d_in[b+2], (const float*)d_in[b+6],
                 (const float*)d_in[b+9], (const float*)d_in[b+8],
                 (const float*)d_in[b+10], (const float*)d_in[b+11], 64, 32 };
    }

    // ---- workspace layout ----
    float* ws = (float*)d_ws;
    size_t off = 0;
    auto alloc = [&](size_t n) { float* p = ws + off; off += n; return p; };
    float* hA    = alloc((size_t)N_NODES * 128);      // current node features
    float* hB    = alloc((size_t)N_NODES * 128);      // next node features
    float* qbuf  = alloc((size_t)N_NODES * 512);
    float* kbuf  = alloc((size_t)N_NODES * 512);
    float* vbuf  = alloc((size_t)N_NODES * 512);
    float* skip  = alloc((size_t)N_NODES * 128);
    float* resid = alloc((size_t)N_NODES * 128);
    float* agg   = alloc((size_t)N_NODES * 512);
    float* alpha = alloc((size_t)N_EDGES * HEADS);
    float* amax  = alloc((size_t)N_NODES * HEADS);
    float* denom = alloc((size_t)N_NODES * HEADS);
    float* s1    = alloc((size_t)N_NODES * 16);
    float* gpool = alloc((size_t)N_GRAPHS * 32);

    const int EH_BLOCKS = 2048, SC_BLOCKS = 4096, TB = 256;

    // ---- input projection: h = relu(x @ in_W + in_b) ----
    launch_gemm(x, in_W, in_b, hA, N_NODES, 2000, 128, 1, stream);

    float* hcur = hA;
    float* hnxt = hB;

    for (int l = 0; l < 3; ++l) {
        const Layer& P = L[l];
        const int C  = P.C;
        const int HC = HEADS * C;

        launch_gemm(hcur, P.Wq, P.bq, qbuf, N_NODES, P.prev, HC, 0, stream);
        launch_gemm(hcur, P.Wk, P.bk, kbuf, N_NODES, P.prev, HC, 0, stream);
        launch_gemm(hcur, P.Wv, P.bv, vbuf, N_NODES, P.prev, HC, 0, stream);
        launch_gemm(hcur, P.Ws, P.bs, skip, N_NODES, P.prev, C, 0, stream);
        const float* residPtr = hcur;
        if (P.rW) {
            launch_gemm(hcur, P.rW, P.rb, resid, N_NODES, P.prev, C, 0, stream);
            residPtr = resid;
        }

        long long nh  = (long long)N_NODES * HEADS;
        long long nhc = (long long)N_NODES * HC;
        fill_f32<<<256, TB, 0, stream>>>(amax,  -INFINITY, nh);
        fill_f32<<<256, TB, 0, stream>>>(denom, 0.0f,      nh);
        fill_f32<<<2048, TB, 0, stream>>>(agg,  0.0f,      nhc);

        const float scale = 1.0f / sqrtf((float)C);
        edge_logits   <<<EH_BLOCKS, TB, 0, stream>>>(qbuf, kbuf, src, dst, alpha, amax, N_EDGES, C, scale);
        edge_expsum   <<<EH_BLOCKS, TB, 0, stream>>>(alpha, amax, denom, dst, N_EDGES);
        edge_normalize<<<EH_BLOCKS, TB, 0, stream>>>(alpha, denom, dst, N_EDGES);
        edge_scatter  <<<SC_BLOCKS, TB, 0, stream>>>(vbuf, alpha, src, dst, agg, N_EDGES, C);

        heads_mean_ln<<<N_NODES, C, 0, stream>>>(agg, skip, residPtr, P.lng, P.lnb, hnxt, C);

        float* t = hcur; hcur = hnxt; hnxt = t;
    }

    // ---- attention pooling ----
    launch_gemm(hcur, att_W1, att_b1, s1, N_NODES, 32, 16, 1, stream);
    fill_f32<<<1, TB, 0, stream>>>(gpool, 0.0f, (long long)N_GRAPHS * 32);
    pool_attention<<<64, TB, 0, stream>>>(s1, att_W2, att_b2, hcur, batch, gpool, 32);

    // ---- classifier ----
    classifier<<<1, 256, 0, stream>>>(gpool, cls_W1, cls_b1, cls_W2, cls_b2,
                                      cls_W3, cls_b3, (float*)d_out);
}
